// GraphConvolution_13589276524974
// MI455X (gfx1250) — matrix operand order repaired and checked by reading, not compiled, not run on hardware
//
#include <hip/hip_runtime.h>
#include <math.h>

// ---------------------------------------------------------------------------
// CDNA5 (gfx1250, wave32) signed-GCN implementation.
// GEMMs use v_wmma_f32_16x16x32_f16 (16x32 tile per wave, A-frag reuse);
// scatter-means use f32 global atomics.
// ---------------------------------------------------------------------------

typedef __attribute__((ext_vector_type(16))) _Float16 v16h;
typedef __attribute__((ext_vector_type(8)))  _Float16 v8h;
typedef __attribute__((ext_vector_type(8)))  float    v8f;

union HFrag { v16h v; v8h h[2]; };

#define D_FEAT 128

// ---- zero fill ------------------------------------------------------------
__global__ void k_zero(float* __restrict__ p, size_t n) {
  size_t i = (size_t)blockIdx.x * blockDim.x + threadIdx.x;
  size_t stride = (size_t)gridDim.x * blockDim.x;
  for (; i < n; i += stride) p[i] = 0.0f;
}

// ---- degree count ---------------------------------------------------------
__global__ void k_deg(const int* __restrict__ dst, int E, float* __restrict__ deg) {
  int i = blockIdx.x * blockDim.x + threadIdx.x;
  if (i < E) atomicAdd(&deg[dst[i]], 1.0f);
}

// ---- invert degree (mean denominator) ------------------------------------
__global__ void k_inv(float* __restrict__ deg, int n) {
  int i = blockIdx.x * blockDim.x + threadIdx.x;
  if (i < n) deg[i] = 1.0f / fmaxf(deg[i], 1.0f);
}

// ---- scatter-add: acc[dst] += feat[src], 4 cols per thread ----------------
__global__ void k_scatter(const float* __restrict__ feat,
                          const int* __restrict__ src,
                          const int* __restrict__ dst,
                          int E, float* __restrict__ acc) {
  size_t i = (size_t)blockIdx.x * blockDim.x + threadIdx.x;
  size_t total = (size_t)E * (D_FEAT / 4);
  if (i >= total) return;
  int e  = (int)(i >> 5);            // D/4 == 32 chunks per edge
  int c4 = ((int)(i & 31)) << 2;
  const float4 v = *(const float4*)(feat + (size_t)src[e] * D_FEAT + c4);
  float* out = acc + (size_t)dst[e] * D_FEAT + c4;
  atomicAdd(out + 0, v.x);
  atomicAdd(out + 1, v.y);
  atomicAdd(out + 2, v.z);
  atomicAdd(out + 3, v.w);
}

// ---- stage [mean*inv , x] -> f16, K = 256 ---------------------------------
__global__ void k_stage2(const float* __restrict__ mean, const float* __restrict__ inv,
                         const float* __restrict__ x, _Float16* __restrict__ A, int n) {
  size_t i = (size_t)blockIdx.x * blockDim.x + threadIdx.x;
  if (i >= (size_t)n * D_FEAT) return;
  int row = (int)(i >> 7);
  int col = (int)(i & 127);
  _Float16* Ar = A + (size_t)row * 256;
  Ar[col]       = (_Float16)(mean[i] * inv[row]);
  Ar[128 + col] = (_Float16)(x[i]);
}

// ---- stage [m0*inv0 , m1*inv1 , h] -> f16, K = 384 ------------------------
__global__ void k_stage3(const float* __restrict__ m0, const float* __restrict__ inv0,
                         const float* __restrict__ m1, const float* __restrict__ inv1,
                         const float* __restrict__ h,  _Float16* __restrict__ A, int n) {
  size_t i = (size_t)blockIdx.x * blockDim.x + threadIdx.x;
  if (i >= (size_t)n * D_FEAT) return;
  int row = (int)(i >> 7);
  int col = (int)(i & 127);
  _Float16* Ar = A + (size_t)row * 384;
  Ar[col]       = (_Float16)(m0[i] * inv0[row]);
  Ar[128 + col] = (_Float16)(m1[i] * inv1[row]);
  Ar[256 + col] = (_Float16)(h[i]);
}

// ---- pack W (K x Nout, f32, row-major) -> Wt (Nout x K, f16) --------------
__global__ void k_packW(const float* __restrict__ W, _Float16* __restrict__ Wt,
                        int K, int Nout) {
  int i = blockIdx.x * blockDim.x + threadIdx.x;
  if (i >= K * Nout) return;
  int k = i / Nout;
  int n = i - k * Nout;
  Wt[(size_t)n * K + k] = (_Float16)W[i];
}

// ---- WMMA GEMM + bias + tanh; optional f32 and/or f16 output --------------
// A:  nrows x K, f16 row-major.     Wt: Nout x K, f16 row-major (transposed W).
// out = tanh(A @ W + bias). 8 waves per block; each wave owns a 16x32 tile
// (two accumulators, A-fragment reused). Block covers 128 rows x 32 cols.
__global__ void __launch_bounds__(256)
k_gemm(const _Float16* __restrict__ A, const _Float16* __restrict__ Wt,
       const float* __restrict__ bias, float* __restrict__ outF,
       _Float16* __restrict__ outH, int nrows, int K, int Nout,
       int ldH, int colOff) {
  const int lane   = threadIdx.x & 31;
  const int wave   = threadIdx.x >> 5;
  const int mBase  = (blockIdx.y * 8 + wave) * 16;
  const int nBase  = blockIdx.x * 32;
  if (mBase >= nrows) return;                 // wave-uniform: EXEC stays all-ones

  const int half16 = lane >> 4;               // 0: lanes 0-15, 1: lanes 16-31
  const int l16    = lane & 15;
  int mRow = mBase + l16;
  if (mRow >= nrows) mRow = nrows - 1;        // clamp (stores masked later)

  // A fragment: lane<16 holds K 0..15; lane>=16 holds K 16..31
  const _Float16* Arow  = A + (size_t)mRow * K + half16 * 16;
  // B fragments: lane<16 -> K 0..15 of its column; lane>=16 -> K 16..31
  const _Float16* Brow0 = Wt + (size_t)(nBase + l16) * K + half16 * 16;
  const _Float16* Brow1 = Brow0 + (size_t)16 * K;

  v8f acc0 = {};
  v8f acc1 = {};
  for (int k0 = 0; k0 < K; k0 += 32) {
    HFrag a, b0, b1;
    a.h[0]  = *(const v8h*)(Arow  + k0);
    a.h[1]  = *(const v8h*)(Arow  + k0 + 8);
    b0.h[0] = *(const v8h*)(Brow0 + k0);
    b0.h[1] = *(const v8h*)(Brow0 + k0 + 8);
    b1.h[0] = *(const v8h*)(Brow1 + k0);
    b1.h[1] = *(const v8h*)(Brow1 + k0 + 8);
    acc0 = __builtin_amdgcn_wmma_f32_16x16x32_f16(
        false, a.v, false, b0.v, (short)0, acc0, false, false);
    acc1 = __builtin_amdgcn_wmma_f32_16x16x32_f16(
        false, a.v, false, b1.v, (short)0, acc1, false, false);
  }

  // C/D layout: VGPR r -> row (mBase + 8*half16 + r), col = nBase + l16
  const int col0  = nBase + l16;
  const int col1  = col0 + 16;
  const float bv0 = bias[col0];
  const float bv1 = bias[col1];
  const int rBase = mBase + half16 * 8;
#pragma unroll
  for (int r = 0; r < 8; ++r) {
    int row = rBase + r;
    if (row < nrows) {
      float v0 = tanhf(acc0[r] + bv0);
      float v1 = tanhf(acc1[r] + bv1);
      if (outF) {
        float* o = outF + (size_t)row * Nout;
        o[col0] = v0;
        o[col1] = v1;
      }
      if (outH) {
        _Float16* o = outH + (size_t)row * ldH + colOff;
        o[col0] = (_Float16)v0;
        o[col1] = (_Float16)v1;
      }
    }
  }
}

// ---------------------------------------------------------------------------
extern "C" void kernel_launch(void* const* d_in, const int* in_sizes, int n_in,
                              void* d_out, int out_size, void* d_ws, size_t ws_size,
                              hipStream_t stream) {
  const float* x       = (const float*)d_in[0];
  const int*   pos_src = (const int*)d_in[1];
  const int*   pos_dst = (const int*)d_in[2];
  const int*   neg_src = (const int*)d_in[3];
  const int*   neg_dst = (const int*)d_in[4];
  const float* W_oneB  = (const float*)d_in[5];
  const float* b_oneB  = (const float*)d_in[6];
  const float* W_oneH  = (const float*)d_in[7];
  const float* b_oneH  = (const float*)d_in[8];
  const float* W_twoB  = (const float*)d_in[9];
  const float* b_twoB  = (const float*)d_in[10];
  const float* W_twoH  = (const float*)d_in[11];
  const float* b_twoH  = (const float*)d_in[12];
  const float* W_four  = (const float*)d_in[13];
  const float* b_four  = (const float*)d_in[14];

  const int N = in_sizes[0] / D_FEAT;
  const int E = in_sizes[1];
  const size_t NF = (size_t)N * D_FEAT;

  // ---- workspace carve-up (256B aligned regions) ----
  auto alignUp = [](size_t v) { return (v + 255) & ~(size_t)255; };
  char* w = (char*)d_ws;
  const size_t nB   = alignUp((size_t)N * sizeof(float));
  const size_t fB   = alignUp(NF * sizeof(float));
  const size_t aB3  = alignUp((size_t)N * 384 * sizeof(_Float16));
  const size_t aB4  = alignUp((size_t)N * 512 * sizeof(_Float16));
  float*    posInv = (float*)w;            w += nB;
  float*    negInv = (float*)w;            w += nB;
  float*    R0 = (float*)w;                w += fB;   // pos1 / mbp
  float*    R1 = (float*)w;                w += fB;   // neg1 / mnn
  float*    R2 = (float*)w;                w += fB;   // mnp
  float*    R3 = (float*)w;                w += fB;   // mbn
  float*    R4 = (float*)w;                w += fB;   // h_b1
  float*    R5 = (float*)w;                w += fB;   // h_n1
  _Float16* A16  = (_Float16*)w;           w += aB3;  // staged GEMM A (<=N x 384)
  _Float16* A16f = (_Float16*)w;           w += aB4;  // [h_b2, h_n2] f16 (N x 512)
  _Float16* Wt   = (_Float16*)w;                      // packed weights (<=256x512)

  const int BT = 256;
  auto blocks = [&](size_t n) { return (unsigned)((n + BT - 1) / BT); };

  auto gemm = [&](const _Float16* Amat, const float* Wf32, const float* bias,
                  float* outF, _Float16* outH, int K, int Nout,
                  int ldH, int colOff) {
    k_packW<<<blocks((size_t)K * Nout), BT, 0, stream>>>(Wf32, Wt, K, Nout);
    dim3 g(Nout / 32, (N + 127) / 128);
    k_gemm<<<g, BT, 0, stream>>>(Amat, Wt, bias, outF, outH, N, K, Nout, ldH, colOff);
  };

  // ---- degrees (shared by both layers) ----
  k_zero<<<blocks(N), BT, 0, stream>>>(posInv, (size_t)N);
  k_zero<<<blocks(N), BT, 0, stream>>>(negInv, (size_t)N);
  k_deg<<<blocks(E), BT, 0, stream>>>(pos_dst, E, posInv);
  k_deg<<<blocks(E), BT, 0, stream>>>(neg_dst, E, negInv);
  k_inv<<<blocks(N), BT, 0, stream>>>(posInv, N);
  k_inv<<<blocks(N), BT, 0, stream>>>(negInv, N);

  // ---- layer 1: seg_mean(x) over pos / neg edges ----
  k_zero<<<blocks(NF), BT, 0, stream>>>(R0, NF);
  k_zero<<<blocks(NF), BT, 0, stream>>>(R1, NF);
  const size_t SC = (size_t)E * (D_FEAT / 4);
  k_scatter<<<blocks(SC), BT, 0, stream>>>(x, pos_src, pos_dst, E, R0);
  k_scatter<<<blocks(SC), BT, 0, stream>>>(x, neg_src, neg_dst, E, R1);

  // h_b1 = tanh([pos1, x] @ W_oneB + b)
  k_stage2<<<blocks(NF), BT, 0, stream>>>(R0, posInv, x, A16, N);
  gemm(A16, W_oneB, b_oneB, R4, nullptr, 256, 128, 0, 0);
  // h_n1 = tanh([neg1, x] @ W_oneH + b)
  k_stage2<<<blocks(NF), BT, 0, stream>>>(R1, negInv, x, A16, N);
  gemm(A16, W_oneH, b_oneH, R5, nullptr, 256, 128, 0, 0);

  // ---- layer 2: four seg_means over h_b1 / h_n1 ----
  k_zero<<<blocks(NF), BT, 0, stream>>>(R0, NF);
  k_zero<<<blocks(NF), BT, 0, stream>>>(R1, NF);
  k_zero<<<blocks(NF), BT, 0, stream>>>(R2, NF);
  k_zero<<<blocks(NF), BT, 0, stream>>>(R3, NF);
  k_scatter<<<blocks(SC), BT, 0, stream>>>(R4, pos_src, pos_dst, E, R0); // mean_pos(h_b1)
  k_scatter<<<blocks(SC), BT, 0, stream>>>(R5, neg_src, neg_dst, E, R1); // mean_neg(h_n1)
  k_scatter<<<blocks(SC), BT, 0, stream>>>(R5, pos_src, pos_dst, E, R2); // mean_pos(h_n1)
  k_scatter<<<blocks(SC), BT, 0, stream>>>(R4, neg_src, neg_dst, E, R3); // mean_neg(h_b1)

  // h_b2 (f16, cols 0..255 of A16f) = tanh([mbp, mnn, h_b1] @ W_twoB + b)
  k_stage3<<<blocks(NF), BT, 0, stream>>>(R0, posInv, R1, negInv, R4, A16, N);
  gemm(A16, W_twoB, b_twoB, nullptr, A16f, 384, 256, 512, 0);
  // h_n2 (f16, cols 256..511) = tanh([mnp, mbn, h_n1] @ W_twoH + b)
  k_stage3<<<blocks(NF), BT, 0, stream>>>(R2, posInv, R3, negInv, R5, A16, N);
  gemm(A16, W_twoH, b_twoH, nullptr, A16f, 384, 256, 512, 256);

  // final: out = tanh([h_b2, h_n2] @ W_four + b)  (f32 to d_out)
  gemm(A16f, W_four, b_four, (float*)d_out, nullptr, 512, 128, 0, 0);
}
